// MultiHeadAttentionLayer_36507222016496
// MI455X (gfx1250) — matrix operand order repaired and checked by reading, not compiled, not run on hardware
//
#include <hip/hip_runtime.h>
#include <cmath>

// ---------------------------------------------------------------------------
// Problem constants (match reference)
// ---------------------------------------------------------------------------
#define Bn 4
#define Sn 1024
#define Dn 512
#define Hn 8
#define HDn 64          // head dim
#define NEGV (-1e10f)

typedef __attribute__((ext_vector_type(16))) __bf16 v16bf;
typedef __attribute__((ext_vector_type(8)))  float  v8f;

// ---------------------------------------------------------------------------
// WMMA fragment loaders for v_wmma_f32_16x16x32_bf16 (wave32).
// Layouts used here (contiguous k order, shared by A and B):
//   A (16x32 bf16): h=lane>>4, m=lane&15.
//     VGPR j: K = 16h + 2j (same k order as B)
//   B (32x16 bf16): h=lane>>4, n=lane&15. VGPR j: K = 16h + 2j at column n
//   C/D (16x16 f32): VGPR r holds element (m = r + 8h, n = lane&15)
// ---------------------------------------------------------------------------
__device__ __forceinline__ v16bf ldA_f32(const float* p, int ld, int lane) {
  const int h = lane >> 4, m = lane & 15;
  v16bf a;
#pragma unroll
  for (int j = 0; j < 8; ++j) {
    const int k = 16 * h + 2 * j;
    a[2 * j]     = (__bf16)p[m * ld + k];
    a[2 * j + 1] = (__bf16)p[m * ld + k + 1];
  }
  return a;
}

__device__ __forceinline__ v16bf ldA_bf16(const __bf16* p, int ld, int lane) {
  const int h = lane >> 4, m = lane & 15;
  v16bf a;
#pragma unroll
  for (int j = 0; j < 8; ++j) {
    const int k = 16 * h + 2 * j;
    a[2 * j]     = p[m * ld + k];   // pairs contiguous -> merged b32 loads
    a[2 * j + 1] = p[m * ld + k + 1];
  }
  return a;
}

__device__ __forceinline__ v16bf ldB_f32(const float* p, int ld, int lane) {
  const int h = lane >> 4, n = lane & 15;
  v16bf b;
#pragma unroll
  for (int j = 0; j < 8; ++j) {
    const int k = 16 * h + 2 * j;
    b[2 * j]     = (__bf16)p[k * ld + n];
    b[2 * j + 1] = (__bf16)p[(k + 1) * ld + n];
  }
  return b;
}

// B fragment gathered transposed: logical B[k][n] = p[n*ld + k]
// (pairs contiguous in k -> merged b32 loads)
__device__ __forceinline__ v16bf ldBT_bf16(const __bf16* p, int ld, int lane) {
  const int h = lane >> 4, n = lane & 15;
  v16bf b;
#pragma unroll
  for (int j = 0; j < 8; ++j) {
    const int k = 16 * h + 2 * j;
    b[2 * j]     = p[n * ld + k];
    b[2 * j + 1] = p[n * ld + k + 1];
  }
  return b;
}

#define WMMA_BF16(acc, a, b) \
  __builtin_amdgcn_wmma_f32_16x16x32_bf16(false, (a), false, (b), (short)0, (acc), false, false)

// ---------------------------------------------------------------------------
// 1) Fused QKV projection:  [B*S, D] x [D, D] + bias
//    One wave -> 16 x 64 output tile (4 accumulators, A-fragment reused 4x).
//    The 64-col group == one head, so Q/K go out as bf16 [B][H][S][HD] and
//    V goes out transposed as bf16 [B][H][HD][S] (for contiguous AV B-frags).
//    grid = (256, 8, 3); block = 32
// ---------------------------------------------------------------------------
__global__ __launch_bounds__(32) void mha_proj_kernel(
    const float* __restrict__ q_in, const float* __restrict__ k_in,
    const float* __restrict__ v_in,
    const float* __restrict__ Wq, const float* __restrict__ bq,
    const float* __restrict__ Wk, const float* __restrict__ bk,
    const float* __restrict__ Wv, const float* __restrict__ bv,
    __bf16* __restrict__ Qws, __bf16* __restrict__ Kws, __bf16* __restrict__ Vws)
{
  const int lane = threadIdx.x;
  const int tm = blockIdx.x;            // 16-row tile over B*S
  const int tn = blockIdx.y;            // 64-col group == head index
  const int which = blockIdx.z;         // 0=Q 1=K 2=V

  const float* X    = (which == 0) ? q_in : (which == 1) ? k_in : v_in;
  const float* W    = (which == 0) ? Wq   : (which == 1) ? Wk   : Wv;
  const float* bias = (which == 0) ? bq   : (which == 1) ? bk   : bv;
  __bf16*      O    = (which == 0) ? Qws  : (which == 1) ? Kws  : Vws;

  const float* Abase = X + (size_t)tm * 16 * Dn;
  const float* Bbase = W + tn * 64;

  v8f acc0 = {}, acc1 = {}, acc2 = {}, acc3 = {};
#pragma unroll 1
  for (int k0 = 0; k0 < Dn; k0 += 32) {
    v16bf a  = ldA_f32(Abase + k0, Dn, lane);
    const float* Bp = Bbase + (size_t)k0 * Dn;
    v16bf b0 = ldB_f32(Bp +  0, Dn, lane);
    v16bf b1 = ldB_f32(Bp + 16, Dn, lane);
    v16bf b2 = ldB_f32(Bp + 32, Dn, lane);
    v16bf b3 = ldB_f32(Bp + 48, Dn, lane);
    acc0 = WMMA_BF16(acc0, a, b0);
    acc1 = WMMA_BF16(acc1, a, b1);
    acc2 = WMMA_BF16(acc2, a, b2);
    acc3 = WMMA_BF16(acc3, a, b3);
  }

  const int h = lane >> 4, nl = lane & 15;
  const int head = tn;
  v8f accs[4] = {acc0, acc1, acc2, acc3};
#pragma unroll
  for (int c = 0; c < 4; ++c) {
    const int hd = 16 * c + nl;
    const float bv_ = bias[tn * 64 + hd];
#pragma unroll
    for (int r = 0; r < 8; ++r) {
      const int mg = tm * 16 + r + 8 * h; // row in B*S
      const int b_ = mg >> 10;            // / S
      const int s_ = mg & 1023;
      const __bf16 val = (__bf16)(accs[c][r] + bv_);
      if (which == 2)                     // V transposed: [B][H][HD][S]
        O[(((size_t)b_ * Hn + head) * HDn + hd) * Sn + s_] = val;
      else                                // Q/K: [B][H][S][HD]
        O[(((size_t)b_ * Hn + head) * Sn + s_) * HDn + hd] = val;
    }
  }
}

// ---------------------------------------------------------------------------
// 2) Continuous-position-bias table: only 63x63 distinct (dr,dc) pairs.
//    tab[(dr+31)*63 + (dc+31)][h]
// ---------------------------------------------------------------------------
__global__ void mha_bias_table_kernel(
    const float* __restrict__ W1, const float* __restrict__ b1,
    const float* __restrict__ W2, const float* __restrict__ b2,
    float* __restrict__ tab)
{
  const int idx = blockIdx.x * blockDim.x + threadIdx.x;
  if (idx >= 63 * 63) return;
  const int dr = idx / 63 - 31;
  const int dc = idx % 63 - 31;
  const float r0 = ((dr > 0) ? 1.f : (dr < 0) ? -1.f : 0.f) * log1pf(fabsf((float)dr));
  const float r1 = ((dc > 0) ? 1.f : (dc < 0) ? -1.f : 0.f) * log1pf(fabsf((float)dc));
  float o[Hn];
#pragma unroll
  for (int hh = 0; hh < Hn; ++hh) o[hh] = b2[hh];
  for (int j = 0; j < 128; ++j) {
    float hv = fmaxf(r0 * W1[j] + r1 * W1[128 + j] + b1[j], 0.f);
#pragma unroll
    for (int hh = 0; hh < Hn; ++hh) o[hh] += hv * W2[j * Hn + hh];
  }
#pragma unroll
  for (int hh = 0; hh < Hn; ++hh) tab[(size_t)idx * Hn + hh] = o[hh];
}

// ---------------------------------------------------------------------------
// 3) FUSED energy + softmax.
//    One 256-thread block (8 waves) owns a 16-row q-stripe. The full
//    16 x 1024 f32 logit stripe lives in LDS (64 KB of the WGP's 320 KB).
//    Each wave computes 8 k-tiles with WMMA (Q fragments loaded once),
//    softmax runs in LDS with __shfl_xor row reductions, and the
//    *normalized* attention is written to d_out exactly once.
//    grid = (64, 32); block = 256
// ---------------------------------------------------------------------------
__global__ __launch_bounds__(256) void mha_energy_softmax_kernel(
    const __bf16* __restrict__ Qws, const __bf16* __restrict__ Kws,
    const float* __restrict__ bias_tab, const long long* __restrict__ mask,
    float* __restrict__ att)
{
  __shared__ float tile[16 * 1024];     // 64 KB logit stripe

  const int t = threadIdx.x;
  const int lane = t & 31;
  const int w = t >> 5;                 // wave id 0..7
  const int qt = blockIdx.x;            // q tile (16 rows)
  const int bh = blockIdx.y;
  const int b = bh >> 3, hidx = bh & 7;

  const __bf16* Qh    = Qws + (size_t)bh * Sn * HDn + (size_t)qt * 16 * HDn;
  const __bf16* Kbase = Kws + (size_t)bh * Sn * HDn;

  // Q fragments: loaded once, reused for all 8 k-tiles of this wave
  v16bf a0 = ldA_bf16(Qh,      HDn, lane);
  v16bf a1 = ldA_bf16(Qh + 32, HDn, lane);

  const int h = lane >> 4, nl = lane & 15;
#pragma unroll 1
  for (int kt = w * 8; kt < w * 8 + 8; ++kt) {
    const __bf16* Kh = Kbase + (size_t)kt * 16 * HDn;
    v16bf b0 = ldBT_bf16(Kh,      HDn, lane);
    v16bf b1 = ldBT_bf16(Kh + 32, HDn, lane);
    v8f acc = {};
    acc = WMMA_BF16(acc, a0, b0);
    acc = WMMA_BF16(acc, a1, b1);

    const int kcol = kt * 16 + nl;
    const long long mv = mask[(size_t)b * Sn + kcol];
    const int rk = kcol >> 5, ck = kcol & 31;
#pragma unroll
    for (int r = 0; r < 8; ++r) {
      const int m = r + 8 * h;          // row within stripe
      const int qrow = qt * 16 + m;
      const int rq = qrow >> 5, cq = qrow & 31;
      float e = acc[r] * 0.125f
              + bias_tab[((rq - rk + 31) * 63 + (cq - ck + 31)) * Hn + hidx];
      if (mv == 0) e = NEGV;
      tile[m * 1024 + kcol] = e;
    }
  }
  __syncthreads();

  // Softmax: 16 threads per row (a 16-lane shuffle group), 64 contiguous
  // columns per thread. Row r threads are lanes [0..15] or [16..31] of one
  // wave, so width-16 __shfl_xor reduces within the row.
  const int r = t >> 4, j = t & 15;
  float* rowp = tile + r * 1024 + j * 64;

  float lmax = -INFINITY;
#pragma unroll
  for (int i = 0; i < 64; ++i) lmax = fmaxf(lmax, rowp[i]);
#pragma unroll
  for (int off = 8; off >= 1; off >>= 1)
    lmax = fmaxf(lmax, __shfl_xor(lmax, off, 16));

  float lsum = 0.f;
#pragma unroll
  for (int i = 0; i < 64; ++i) {
    const float e = __expf(rowp[i] - lmax);
    rowp[i] = e;
    lsum += e;
  }
#pragma unroll
  for (int off = 8; off >= 1; off >>= 1)
    lsum += __shfl_xor(lsum, off, 16);
  const float inv = 1.0f / lsum;

  float* gp = att + ((size_t)bh * Sn + qt * 16 + r) * Sn + j * 64;
#pragma unroll
  for (int i = 0; i < 64; i += 4) {
    float4 v4;
    v4.x = rowp[i]     * inv; v4.y = rowp[i + 1] * inv;
    v4.z = rowp[i + 2] * inv; v4.w = rowp[i + 3] * inv;
    *(float4*)(gp + i) = v4;
  }
}

// ---------------------------------------------------------------------------
// 4) x = A · V  ->  bf16 xtmp [B][S][D]
//    One wave -> 16 x 64 tile (all of HD, 4 accumulators; A-frag reused 4x).
//    V is stored transposed [B][H][HD][S] so B-frags gather contiguous pairs.
//    Streaming attention reads are prefetched (global_prefetch_b8).
//    grid = (64, 32); block = 32
// ---------------------------------------------------------------------------
__global__ __launch_bounds__(32) void mha_av_kernel(
    const float* __restrict__ att, const __bf16* __restrict__ Vws,
    __bf16* __restrict__ xtmp)
{
  const int lane = threadIdx.x;
  const int qt = blockIdx.x, bh = blockIdx.y;
  const int b = bh >> 3, hidx = bh & 7;

  const float*  Ab = att + ((size_t)bh * Sn + (size_t)qt * 16) * Sn;
  const __bf16* Vt = Vws + (size_t)bh * HDn * Sn;   // [HD][S]

  v8f acc0 = {}, acc1 = {}, acc2 = {}, acc3 = {};
#pragma unroll 1
  for (int k0 = 0; k0 < Sn; k0 += 32) {
    __builtin_prefetch(Ab + (size_t)(lane & 15) * Sn + k0 + 128, 0, 1);
    v16bf a  = ldA_f32(Ab + k0, Sn, lane);
    v16bf b0 = ldBT_bf16(Vt + (size_t)(0)  * Sn + k0, Sn, lane);
    v16bf b1 = ldBT_bf16(Vt + (size_t)(16) * Sn + k0, Sn, lane);
    v16bf b2 = ldBT_bf16(Vt + (size_t)(32) * Sn + k0, Sn, lane);
    v16bf b3 = ldBT_bf16(Vt + (size_t)(48) * Sn + k0, Sn, lane);
    acc0 = WMMA_BF16(acc0, a, b0);
    acc1 = WMMA_BF16(acc1, a, b1);
    acc2 = WMMA_BF16(acc2, a, b2);
    acc3 = WMMA_BF16(acc3, a, b3);
  }

  const int h = lane >> 4, nl = lane & 15;
  v8f accs[4] = {acc0, acc1, acc2, acc3};
#pragma unroll
  for (int c = 0; c < 4; ++c) {
    const int dcol = hidx * HDn + 16 * c + nl;
#pragma unroll
    for (int r = 0; r < 8; ++r) {
      const int qrow = qt * 16 + r + 8 * h;
      xtmp[((size_t)b * Sn + qrow) * Dn + dcol] = (__bf16)accs[c][r];
    }
  }
}

// ---------------------------------------------------------------------------
// 5) out = xtmp · Wo + bo  -> f32 x slice of d_out
//    One wave -> 16 x 64 tile. grid = (256, 8); block = 32
// ---------------------------------------------------------------------------
__global__ __launch_bounds__(32) void mha_oproj_kernel(
    const __bf16* __restrict__ xtmp, const float* __restrict__ Wo,
    const float* __restrict__ bo, float* __restrict__ out)
{
  const int lane = threadIdx.x;
  const int tm = blockIdx.x, tn = blockIdx.y;

  const __bf16* Abase = xtmp + (size_t)tm * 16 * Dn;
  const float*  Bbase = Wo + tn * 64;

  v8f acc0 = {}, acc1 = {}, acc2 = {}, acc3 = {};
#pragma unroll 1
  for (int k0 = 0; k0 < Dn; k0 += 32) {
    v16bf a  = ldA_bf16(Abase + k0, Dn, lane);
    const float* Bp = Bbase + (size_t)k0 * Dn;
    v16bf b0 = ldB_f32(Bp +  0, Dn, lane);
    v16bf b1 = ldB_f32(Bp + 16, Dn, lane);
    v16bf b2 = ldB_f32(Bp + 32, Dn, lane);
    v16bf b3 = ldB_f32(Bp + 48, Dn, lane);
    acc0 = WMMA_BF16(acc0, a, b0);
    acc1 = WMMA_BF16(acc1, a, b1);
    acc2 = WMMA_BF16(acc2, a, b2);
    acc3 = WMMA_BF16(acc3, a, b3);
  }

  const int h = lane >> 4, nl = lane & 15;
  v8f accs[4] = {acc0, acc1, acc2, acc3};
#pragma unroll
  for (int c = 0; c < 4; ++c) {
    const int ng = tn * 64 + 16 * c + nl;
    const float bb = bo[ng];
#pragma unroll
    for (int r = 0; r < 8; ++r) {
      const int mg = tm * 16 + r + 8 * h;
      out[(size_t)mg * Dn + ng] = accs[c][r] + bb;
    }
  }
}

// ---------------------------------------------------------------------------
// Launch
// ---------------------------------------------------------------------------
extern "C" void kernel_launch(void* const* d_in, const int* in_sizes, int n_in,
                              void* d_out, int out_size, void* d_ws, size_t ws_size,
                              hipStream_t stream) {
  (void)in_sizes; (void)n_in; (void)out_size; (void)ws_size;

  const float*     query = (const float*)d_in[0];
  const float*     key_  = (const float*)d_in[1];
  const float*     value = (const float*)d_in[2];
  const long long* mask  = (const long long*)d_in[3];
  const float* Wq = (const float*)d_in[4];  const float* bq = (const float*)d_in[5];
  const float* Wk = (const float*)d_in[6];  const float* bk = (const float*)d_in[7];
  const float* Wv = (const float*)d_in[8];  const float* bv = (const float*)d_in[9];
  const float* Wo = (const float*)d_in[10]; const float* bo = (const float*)d_in[11];
  const float* W1 = (const float*)d_in[12]; const float* b1 = (const float*)d_in[13];
  const float* W2 = (const float*)d_in[14]; const float* b2 = (const float*)d_in[15];

  float* out  = (float*)d_out;
  float* xout = out;                                  // [B,S,D]
  float* att  = out + (size_t)Bn * Sn * Dn;           // [B,H,S,S]

  // workspace layout
  const size_t QKV_BYTES = (size_t)Bn * Hn * Sn * HDn * sizeof(__bf16); // 4 MB each
  char* ws = (char*)d_ws;
  __bf16* Qws  = (__bf16*)(ws);                       // [B,H,S,HD]
  __bf16* Kws  = (__bf16*)(ws + QKV_BYTES);           // [B,H,S,HD]
  __bf16* Vws  = (__bf16*)(ws + 2 * QKV_BYTES);       // [B,H,HD,S] (transposed)
  __bf16* xtmp = (__bf16*)(ws + 3 * QKV_BYTES);       // [B,S,D]
  float*  btab = (float*)(ws + 4 * QKV_BYTES);        // 63*63*H f32

  // 1) QKV projections (bf16 WMMA, f32 accumulate, 16x64 tiles)
  mha_proj_kernel<<<dim3((Bn * Sn) / 16, Hn, 3), 32, 0, stream>>>(
      query, key_, value, Wq, bq, Wk, bk, Wv, bv, Qws, Kws, Vws);

  // 2) 63x63xH bias table (replaces the S^2 MLP)
  mha_bias_table_kernel<<<(63 * 63 + 255) / 256, 256, 0, stream>>>(
      W1, b1, W2, b2, btab);

  // 3) fused logits + softmax -> normalized attention written once
  mha_energy_softmax_kernel<<<dim3(Sn / 16, Bn * Hn), 256, 0, stream>>>(
      Qws, Kws, btab, mask, att);

  // 4) A·V -> bf16 xtmp (16x64 tiles, prefetched streaming reads)
  mha_av_kernel<<<dim3(Sn / 16, Bn * Hn), 32, 0, stream>>>(
      att, Vws, xtmp);

  // 5) output projection -> x slice of d_out
  mha_oproj_kernel<<<dim3((Bn * Sn) / 16, Hn), 32, 0, stream>>>(
      xtmp, Wo, bo, xout);
}
